// SOM_61753039782108
// MI455X (gfx1250) — compile-verified, matrix operand order checked by reading, not run on hardware
//
#include <hip/hip_runtime.h>
#include <stdint.h>

#define D_FEAT 784
#define KP     800       // K padded to multiple of 32 (bf16 zeros are inert)
#define B_ROWS 8192
#define N_W    10000
#define BM     128
#define BN     128
#define BK     32
#define N_WP   10112     // N padded to multiple of BN (sentinel w_sq = 1e30)
#define LDK    40        // LDS row stride in bf16 elems (80B; breaks bank pattern)
#define NT     (KP / BK) // 25 K-steps

#define AS1 __attribute__((address_space(1)))
#define AS3 __attribute__((address_space(3)))

typedef __attribute__((ext_vector_type(16))) __bf16        v16bf;
typedef __attribute__((ext_vector_type(8)))  float         v8f;
typedef __attribute__((ext_vector_type(4)))  unsigned int  u32x4;
typedef int v4i_vs __attribute__((vector_size(4 * sizeof(int))));  // matches builtin param

union Frag { v16bf v; u32x4 q[2]; };

#if __has_builtin(__builtin_amdgcn_global_load_async_to_lds_b128)
#define HAVE_ASYNC_LDS 1
__device__ __forceinline__ void async_cp16(const unsigned short* g, unsigned short* l) {
    __builtin_amdgcn_global_load_async_to_lds_b128(
        (AS1 v4i_vs*)(uintptr_t)g,
        (AS3 v4i_vs*)(unsigned int)(uintptr_t)l,
        /*offset=*/0, /*cpol=*/0);
}
#endif

__device__ __forceinline__ unsigned short f2bf(float f) {
    unsigned int u = __float_as_uint(f);
    u += 0x7FFFu + ((u >> 16) & 1u);   // round-to-nearest-even
    return (unsigned short)(u >> 16);
}

// One block per (padded) row: convert to bf16 with K-padding, compute ||row||^2.
__global__ void som_prep(const float* __restrict__ src,
                         unsigned short* __restrict__ dst,
                         float* __restrict__ sq, int nreal) {
    const int row = blockIdx.x;
    const int tid = threadIdx.x;
    __shared__ float red[256];
    float s = 0.0f;
    if (row < nreal) {
        const float* r = src + (size_t)row * D_FEAT;
        for (int k = tid; k < KP; k += 256) {
            float v = (k < D_FEAT) ? r[k] : 0.0f;
            dst[(size_t)row * KP + k] = f2bf(v);
            s += v * v;
        }
    } else {
        for (int k = tid; k < KP; k += 256) dst[(size_t)row * KP + k] = 0;
        s = (tid == 0) ? 1e30f : 0.0f;   // padded weight rows never win argmin
    }
    red[tid] = s;
    __syncthreads();
    for (int off = 128; off > 0; off >>= 1) {
        if (tid < off) red[tid] += red[tid + off];
        __syncthreads();
    }
    if (tid == 0) sq[row] = red[0];
}

__global__ void som_init_best(unsigned long long* __restrict__ best) {
    int i = blockIdx.x * blockDim.x + threadIdx.x;
    if (i < B_ROWS) best[i] = ~0ull;
}

// 256 threads = 8 waves. Block tile 128x128, BK=32, double-buffered LDS.
// Wave tile 32(M) x 64(N): 2 A-frags x 4 B-frags -> 8 WMMAs per K-step.
// Epilogue fuses distance + packed-key argmin (u64 atomic min).
__global__ void __launch_bounds__(256)
som_gemm_argmin(const unsigned short* __restrict__ Xbf,
                const unsigned short* __restrict__ Wbf,
                const float* __restrict__ xsq,
                const float* __restrict__ wsq,
                unsigned long long* __restrict__ best) {
    __shared__ unsigned short ldsA[2][BM * LDK];   // 20 KB
    __shared__ unsigned short ldsB[2][BN * LDK];   // 20 KB

    const int tid   = threadIdx.x;
    const int lane  = tid & 31;
    const int wave  = tid >> 5;
    const int l15   = lane & 15;
    const int lhi   = lane >> 4;       // 0/1: which K half-chunks this lane carries
    const int khalf = lhi * 8;

    const int blockM = blockIdx.y * BM;
    const int blockN = blockIdx.x * BN;
    const int waveM  = (wave >> 1) * 32;   // 0,32,64,96
    const int waveN  = (wave & 1) * 64;    // 0,64

    // Cooperative copy: 512 16B chunks for A + 512 for B per K-step;
    // thread handles chunk ids {tid, tid+256} for each of A and B.
    const int q0 = tid, q1 = tid + 256;
    const int r0 = q0 >> 2, s0 = (q0 & 3) * 8;   // row, 16B segment (in halfs)
    const int r1 = q1 >> 2, s1 = (q1 & 3) * 8;
    const unsigned short* gA0 = Xbf + (size_t)(blockM + r0) * KP + s0;
    const unsigned short* gA1 = Xbf + (size_t)(blockM + r1) * KP + s1;
    const unsigned short* gB0 = Wbf + (size_t)(blockN + r0) * KP + s0;
    const unsigned short* gB1 = Wbf + (size_t)(blockN + r1) * KP + s1;
    const int lA0 = r0 * LDK + s0;
    const int lA1 = r1 * LDK + s1;

    v8f acc[2][4] = {};
    Frag af[2];

#ifdef HAVE_ASYNC_LDS
    // ---- async global -> LDS path (ASYNCcnt-tracked, no VGPR staging) ----
    {
        async_cp16(gA0, &ldsA[0][lA0]);
        async_cp16(gA1, &ldsA[0][lA1]);
        async_cp16(gB0, &ldsB[0][lA0]);
        async_cp16(gB1, &ldsB[0][lA1]);
    }
    int cur = 0;
    for (int kt = 0; kt < NT; ++kt) {
        asm volatile("s_wait_asynccnt 0x0" ::: "memory"); // my copies landed in LDS
        __syncthreads();                                  // => everyone's copies landed
        if (kt + 1 < NT) {
            const int ko = (kt + 1) * BK;
            async_cp16(gA0 + ko, &ldsA[cur ^ 1][lA0]);
            async_cp16(gA1 + ko, &ldsA[cur ^ 1][lA1]);
            async_cp16(gB0 + ko, &ldsB[cur ^ 1][lA0]);
            async_cp16(gB1 + ko, &ldsB[cur ^ 1][lA1]);
        }
#else
    // ---- fallback: global_load_b128 staging + ds_store_b128 ----
    u32x4 sa0, sa1, sb0, sb1;
    sa0 = *(const u32x4*)gA0; sa1 = *(const u32x4*)gA1;
    sb0 = *(const u32x4*)gB0; sb1 = *(const u32x4*)gB1;
    *(u32x4*)(&ldsA[0][lA0]) = sa0; *(u32x4*)(&ldsA[0][lA1]) = sa1;
    *(u32x4*)(&ldsB[0][lA0]) = sb0; *(u32x4*)(&ldsB[0][lA1]) = sb1;
    int cur = 0;
    for (int kt = 0; kt < NT; ++kt) {
        __syncthreads();
        if (kt + 1 < NT) {
            const int ko = (kt + 1) * BK;
            sa0 = *(const u32x4*)(gA0 + ko); sa1 = *(const u32x4*)(gA1 + ko);
            sb0 = *(const u32x4*)(gB0 + ko); sb1 = *(const u32x4*)(gB1 + ko);
        }
#endif
        // fragment reads (ds_load_b128) + 8 WMMAs on buffer `cur`
#pragma unroll
        for (int i = 0; i < 2; ++i) {
            const unsigned short* pa =
                &ldsA[cur][(waveM + i * 16 + l15) * LDK + khalf];
            af[i].q[0] = *(const u32x4*)pa;
            af[i].q[1] = *(const u32x4*)(pa + 16);
        }
#pragma unroll
        for (int j = 0; j < 4; ++j) {
            Frag b;
            const unsigned short* pb =
                &ldsB[cur][(waveN + j * 16 + l15) * LDK + khalf];
            b.q[0] = *(const u32x4*)pb;
            b.q[1] = *(const u32x4*)(pb + 16);
#pragma unroll
            for (int i = 0; i < 2; ++i)
                acc[i][j] = __builtin_amdgcn_wmma_f32_16x16x32_bf16(
                    false, af[i].v, false, b.v, (short)0, acc[i][j], false, false);
        }
#ifndef HAVE_ASYNC_LDS
        if (kt + 1 < NT) {
            *(u32x4*)(&ldsA[cur ^ 1][lA0]) = sa0;
            *(u32x4*)(&ldsA[cur ^ 1][lA1]) = sa1;
            *(u32x4*)(&ldsB[cur ^ 1][lA0]) = sb0;
            *(u32x4*)(&ldsB[cur ^ 1][lA1]) = sb1;
        }
#endif
        cur ^= 1;
    }

    // Epilogue: C layout lane = N (mod 16), VGPR e = M row (+8 for lanes 16-31).
    float wn[4];
#pragma unroll
    for (int j = 0; j < 4; ++j) wn[j] = wsq[blockN + waveN + j * 16 + l15];
    const int mrow = blockM + waveM + lhi * 8;
#pragma unroll
    for (int i = 0; i < 2; ++i) {
#pragma unroll
        for (int e = 0; e < 8; ++e) {
            const int m = mrow + i * 16 + e;
            const float xs = xsq[m];
            unsigned long long key = ~0ull;
#pragma unroll
            for (int j = 0; j < 4; ++j) {
                float d = fmaxf(xs - 2.0f * acc[i][j][e] + wn[j], 0.0f);
                unsigned long long k =
                    ((unsigned long long)__float_as_uint(d) << 32) |
                    (unsigned int)(blockN + waveN + j * 16 + l15);
                key = (k < key) ? k : key;
            }
#pragma unroll
            for (int s = 1; s < 16; s <<= 1) {   // min across 16 lanes sharing m
                unsigned int hi = __shfl_xor((unsigned int)(key >> 32), s, 32);
                unsigned int lo = __shfl_xor((unsigned int)key, s, 32);
                unsigned long long o = ((unsigned long long)hi << 32) | lo;
                key = (o < key) ? o : key;
            }
            if (l15 == 0) atomicMin(&best[m], key);
        }
    }
}

// out[row] = W[best_index(row)] in f32 (float4 copies; 784*4 bytes % 16 == 0)
__global__ void som_gather(const unsigned long long* __restrict__ best,
                           const float* __restrict__ W,
                           float* __restrict__ out) {
    const int row = blockIdx.x;
    const unsigned int idx = (unsigned int)(best[row] & 0xFFFFFFFFu);
    const float4* src = (const float4*)(W + (size_t)idx * D_FEAT);
    float4*       dst = (float4*)(out + (size_t)row * D_FEAT);
    for (int t = threadIdx.x; t < D_FEAT / 4; t += blockDim.x) dst[t] = src[t];
}

extern "C" void kernel_launch(void* const* d_in, const int* in_sizes, int n_in,
                              void* d_out, int out_size, void* d_ws, size_t ws_size,
                              hipStream_t stream) {
    const float* x = (const float*)d_in[0];   // [8192, 784]
    const float* w = (const float*)d_in[1];   // [10000, 784]
    float* out = (float*)d_out;               // [8192, 784]

    char* ws = (char*)d_ws;
    const size_t offXbf  = 0;
    const size_t offWbf  = offXbf + (size_t)B_ROWS * KP * sizeof(unsigned short);
    const size_t offXsq  = offWbf + (size_t)N_WP   * KP * sizeof(unsigned short);
    const size_t offWsq  = offXsq + (size_t)B_ROWS * sizeof(float);
    const size_t offBest = offWsq + (size_t)N_WP   * sizeof(float);  // 8B aligned
    unsigned short*     Xbf  = (unsigned short*)(ws + offXbf);
    unsigned short*     Wbf  = (unsigned short*)(ws + offWbf);
    float*              Xsq  = (float*)(ws + offXsq);
    float*              Wsq  = (float*)(ws + offWsq);
    unsigned long long* best = (unsigned long long*)(ws + offBest);
    (void)in_sizes; (void)n_in; (void)out_size; (void)ws_size;

    som_prep<<<B_ROWS, 256, 0, stream>>>(x, Xbf, Xsq, B_ROWS);
    som_prep<<<N_WP,   256, 0, stream>>>(w, Wbf, Wsq, N_W);
    som_init_best<<<(B_ROWS + 255) / 256, 256, 0, stream>>>(best);

    dim3 grid(N_WP / BN, B_ROWS / BM);   // 79 x 64 blocks
    som_gemm_argmin<<<grid, 256, 0, stream>>>(Xbf, Wbf, Xsq, Wsq, best);

    som_gather<<<B_ROWS, 256, 0, stream>>>(best, w, out);
}
